// LinearNemotronHMoE_21311627723127
// MI455X (gfx1250) — compile-verified, matrix operand order checked by reading, not run on hardware
//
#include <hip/hip_runtime.h>
#include <math.h>

// ---------------- types ----------------
typedef __attribute__((ext_vector_type(16))) __bf16 v16bf;
typedef __attribute__((ext_vector_type(8)))  float  v8f;

static constexpr int Hdim = 1024;   // hidden size
static constexpr int Idim = 512;    // expert intermediate
static constexpr int ISdim = 1024;  // shared intermediate
static constexpr int XP = 1032;     // LDS pitch (halves) for X tile  (16B aligned, conflict-free)
static constexpr int HP = 520;      // LDS pitch (halves) for hid tile

struct U2 { uint4 a; uint4 b; };
struct B4 { __bf16 h[4]; };

static __device__ inline v8f wmma_bf16(v16bf a, v16bf b, v8f c) {
  // D = A(16x32 bf16) x B(32x16 bf16) + C(16x16 f32)
  return __builtin_amdgcn_wmma_f32_16x16x32_bf16(false, a, false, b, (short)0, c,
                                                 false, false);
}

// bf16 fragment (LDS or global): lane holds K = koff+{0..7} and koff+16+{0..7}
static __device__ inline v16bf frag_bf16(const __bf16* p) {
  U2 u;
  u.a = *(const uint4*)(p);       // 8 halves: K = +0..7
  u.b = *(const uint4*)(p + 16);  // 8 halves: K = +16..23
  return __builtin_bit_cast(v16bf, u);
}

// ---------------- kernel 0: one-shot fp32 -> bf16 weight conversion ----------------
__global__ __launch_bounds__(256)
void cvt_f32_to_bf16(const float* __restrict__ in, __bf16* __restrict__ out, int n4) {
  int i = blockIdx.x * 256 + threadIdx.x;
  if (i < n4) {
    float4 f = ((const float4*)in)[i];
    B4 b;
    b.h[0] = (__bf16)f.x; b.h[1] = (__bf16)f.y;
    b.h[2] = (__bf16)f.z; b.h[3] = (__bf16)f.w;
    *(B4*)(out + (size_t)i * 4) = b;
  }
}

// ---------------- kernel 1: router ----------------
__global__ __launch_bounds__(64)
void moe_router(const float* __restrict__ X, const float* __restrict__ Wg,
                const float* __restrict__ bias, float* __restrict__ cw) {
  const int t = blockIdx.x;
  const int tid = threadIdx.x;
  __shared__ float part[64];
  __shared__ float lg[16];

  const int e = tid & 15;
  const int p = tid >> 4;
  const float* x = X + (size_t)t * Hdim;
  const float* w = Wg + (size_t)e * Hdim;
  float s = 0.f;
  for (int h = p * 256; h < p * 256 + 256; ++h) s += x[h] * w[h];
  part[tid] = s;
  __syncthreads();
  if (tid < 16) lg[tid] = part[tid] + part[tid + 16] + part[tid + 32] + part[tid + 48];
  __syncthreads();

  if (tid == 0) {
    float sc[16], ch[16];
    for (int i = 0; i < 16; ++i) {
      sc[i] = 1.0f / (1.0f + expf(-lg[i]));   // sigmoid scores
      ch[i] = sc[i] + bias[i];                // choice scores
    }
    float gs[4];
    for (int g = 0; g < 4; ++g) {             // per-group sum of top-2
      float m1 = -1e30f, m2 = -1e30f;
      for (int j = 0; j < 4; ++j) {
        float v = ch[g * 4 + j];
        if (v > m1) { m2 = m1; m1 = v; } else if (v > m2) { m2 = v; }
      }
      gs[g] = m1 + m2;
    }
    int g1 = 0; float b1 = -1e30f;            // top-2 groups
    for (int g = 0; g < 4; ++g) if (gs[g] > b1) { b1 = gs[g]; g1 = g; }
    int g2 = 0; float b2 = -1e30f;
    for (int g = 0; g < 4; ++g) if (g != g1 && gs[g] > b2) { b2 = gs[g]; g2 = g; }
    float masked[16];
    for (int i = 0; i < 16; ++i) {
      int g = i >> 2;
      masked[i] = (g == g1 || g == g2) ? ch[i] : 0.0f;
    }
    int idx[4]; float wv[4]; float wsum = 0.f; // top-4 experts
    for (int j = 0; j < 4; ++j) {
      float best = -1e30f; int bi = 0;
      for (int i = 0; i < 16; ++i) if (masked[i] > best) { best = masked[i]; bi = i; }
      idx[j] = bi; masked[bi] = -1e30f;
      wv[j] = sc[bi]; wsum += wv[j];
    }
    const float inv = 2.5f / (wsum + 1e-20f);
    float row[16];
    for (int i = 0; i < 16; ++i) row[i] = 0.f;
    for (int j = 0; j < 4; ++j) row[idx[j]] += wv[j] * inv;
    float* o = cw + (size_t)t * 16;
    for (int i = 0; i < 16; ++i) o[i] = row[i];
  }
}

// ---------------- kernel 2: experts + shared, WMMA bf16 ----------------
__global__ __launch_bounds__(256)
void moe_main(const float* __restrict__ X,
              const __bf16* __restrict__ W1, const __bf16* __restrict__ W2,
              const __bf16* __restrict__ Ws1, const __bf16* __restrict__ Ws2,
              const float* __restrict__ cw, float* __restrict__ out) {
  __shared__ __align__(16) __bf16 Xb[16 * XP];   // 16 tokens x H, bf16
  __shared__ __align__(16) __bf16 Hid[16 * HP];  // 16 tokens x 512, bf16 (cw folded in)
  __shared__ float cwS[16][18];                  // cols 16,17 == 1.0 (shared-expert iters)

  const int tid = threadIdx.x;
  const int lane = tid & 31;
  const int wave = tid >> 5;          // 0..7
  const int tok0 = blockIdx.x * 16;

  const int mrow = lane & 15;         // A: row M / B: col N held by this lane
  const int koff = (lane >> 4) * 8;   // K sub-offset per half-wave
  const int hi8  = (lane >> 4) * 8;   // C/D: M = v + hi8

  // stage X tile as bf16 (coalesced global reads)
  for (int j = tid; j < 16 * Hdim; j += 256) {
    int m = j >> 10, k = j & (Hdim - 1);
    Xb[m * XP + k] = (__bf16)X[(size_t)(tok0 + m) * Hdim + k];
  }
  { int m = tid >> 4, e = tid & 15;
    cwS[m][e] = cw[(size_t)(tok0 + m) * 16 + e];
    if (tid < 32) cwS[tid >> 1][16 + (tid & 1)] = 1.0f; }
  __syncthreads();

  v8f out_acc[8];
  const v8f vzero = {};
#pragma unroll
  for (int nt = 0; nt < 8; ++nt) out_acc[nt] = vzero;

  const int colA = wave * 64 + mrow;    // phase-A output column base for this lane
  const int rowB = wave * 128 + mrow;   // phase-B weight-row base for this lane

  // 18 iterations: 16 routed experts + 2 halves of the shared expert
  for (int it = 0; it < 18; ++it) {
    const bool isExpert = (it < 16);
    const __bf16* wup = isExpert ? (W1 + (size_t)it * Idim * Hdim)
                                 : (Ws1 + (size_t)(it - 16) * 512 * Hdim);
    const __bf16* wdn; size_t dnStride;
    if (isExpert) { wdn = W2 + (size_t)it * Hdim * Idim;  dnStride = Idim; }
    else          { wdn = Ws2 + (size_t)(it - 16) * 512;  dnStride = ISdim; }

    // warm next iteration's up-proj weights (global_prefetch_b8)
    if (it + 1 < 18) {
      const __bf16* nxt = (it + 1 < 16) ? (W1 + (size_t)(it + 1) * Idim * Hdim)
                                        : (Ws1 + (size_t)(it + 1 - 16) * 512 * Hdim);
      __builtin_prefetch(nxt + (size_t)(wave * 64 + mrow) * Hdim, 0, 0);
    }

    // per-lane combine scales for this iteration (1.0 for shared-expert iters)
    float csc[8];
#pragma unroll
    for (int v = 0; v < 8; ++v) csc[v] = cwS[v + hi8][it];

    // ---- phase A: hid = relu^2(X @ Wupᵀ) * cw   (this wave: i-tiles wave*64 .. +63)
    v8f accA[4];
#pragma unroll
    for (int nt = 0; nt < 4; ++nt) accA[nt] = vzero;

    for (int k0 = 0; k0 < Hdim; k0 += 32) {
      v16bf a = frag_bf16(&Xb[mrow * XP + k0 + koff]);
#pragma unroll
      for (int nt = 0; nt < 4; ++nt) {
        v16bf b = frag_bf16(wup + (size_t)(colA + nt * 16) * Hdim + k0 + koff);
        accA[nt] = wmma_bf16(a, b, accA[nt]);
      }
    }
    // branchless epilogue: relu^2, fold cw, store bf16 to LDS
#pragma unroll
    for (int nt = 0; nt < 4; ++nt) {
#pragma unroll
      for (int v = 0; v < 8; ++v) {
        float r = fmaxf(accA[nt][v], 0.0f);
        r = r * r * csc[v];
        Hid[(v + hi8) * HP + colA + nt * 16] = (__bf16)r;
      }
    }
    __syncthreads();

    // ---- phase B: out += hid @ Wdnᵀ   (this wave: h-tiles wave*128 .. +127)
    for (int k0 = 0; k0 < 512; k0 += 32) {
      v16bf a = frag_bf16(&Hid[mrow * HP + k0 + koff]);
#pragma unroll
      for (int nt = 0; nt < 8; ++nt) {
        v16bf b = frag_bf16(wdn + (size_t)(rowB + nt * 16) * dnStride + k0 + koff);
        out_acc[nt] = wmma_bf16(a, b, out_acc[nt]);
      }
    }
    __syncthreads();
  }

  // ---- write output (fp32)
#pragma unroll
  for (int nt = 0; nt < 8; ++nt) {
#pragma unroll
    for (int v = 0; v < 8; ++v) {
      out[(size_t)(tok0 + v + hi8) * Hdim + rowB + nt * 16] = out_acc[nt][v];
    }
  }
}

// ---------------- launch ----------------
extern "C" void kernel_launch(void* const* d_in, const int* in_sizes, int n_in,
                              void* d_out, int out_size, void* d_ws, size_t ws_size,
                              hipStream_t stream) {
  const float* X    = (const float*)d_in[0];  // (B,S,H)
  const float* Wg   = (const float*)d_in[1];  // (E,H)
  const float* bias = (const float*)d_in[2];  // (E,)
  const float* W1f  = (const float*)d_in[3];  // (E,I,H)
  const float* W2f  = (const float*)d_in[4];  // (E,H,I)
  const float* Ws1f = (const float*)d_in[5];  // (IS,H)
  const float* Ws2f = (const float*)d_in[6];  // (H,IS)
  float* out = (float*)d_out;

  const int T = in_sizes[0] / Hdim;           // total tokens (B*S)

  // ---- workspace layout (bytes) ----
  // [0, T*16*4)                : cw (fp32 combine weights)
  // then bf16 copies of W1, W2, Ws1, Ws2 (16B-aligned)
  char* ws = (char*)d_ws;
  float* cw = (float*)ws;
  size_t off = ((size_t)T * 16 * 4 + 255) & ~(size_t)255;
  const size_t n_w1  = (size_t)in_sizes[3];   // 16*512*1024
  const size_t n_w2  = (size_t)in_sizes[4];   // 16*1024*512
  const size_t n_ws1 = (size_t)in_sizes[5];   // 1024*1024
  const size_t n_ws2 = (size_t)in_sizes[6];   // 1024*1024
  __bf16* W1b  = (__bf16*)(ws + off);                  off += n_w1 * 2;
  __bf16* W2b  = (__bf16*)(ws + off);                  off += n_w2 * 2;
  __bf16* Ws1b = (__bf16*)(ws + off);                  off += n_ws1 * 2;
  __bf16* Ws2b = (__bf16*)(ws + off);

  // one-shot weight conversion fp32 -> bf16 (streamed, float4 granularity)
  {
    int n4;
    n4 = (int)(n_w1  / 4); cvt_f32_to_bf16<<<(n4 + 255) / 256, 256, 0, stream>>>(W1f,  W1b,  n4);
    n4 = (int)(n_w2  / 4); cvt_f32_to_bf16<<<(n4 + 255) / 256, 256, 0, stream>>>(W2f,  W2b,  n4);
    n4 = (int)(n_ws1 / 4); cvt_f32_to_bf16<<<(n4 + 255) / 256, 256, 0, stream>>>(Ws1f, Ws1b, n4);
    n4 = (int)(n_ws2 / 4); cvt_f32_to_bf16<<<(n4 + 255) / 256, 256, 0, stream>>>(Ws2f, Ws2b, n4);
  }

  moe_router<<<T, 64, 0, stream>>>(X, Wg, bias, cw);
  moe_main<<<T / 16, 256, 0, stream>>>(X, W1b, W2b, Ws1b, Ws2b, cw, out);
}
